// InfoBottleneckReadout_56349970923947
// MI455X (gfx1250) — matrix operand order repaired and checked by reading, not compiled, not run on hardware
//
#include <hip/hip_runtime.h>
#include <hip/hip_bf16.h>
#include <math.h>

#define B_   4
#define N_   512
#define P_   2048
#define E_   32768
#define F_   128
#define DRBF 32
#define INDIM 416            // DRBF + 3*F_
#define CUT  4.0f
#define PI_  3.14159265358979f

typedef __attribute__((ext_vector_type(16))) __bf16 v16bf;
typedef __attribute__((ext_vector_type(8)))  __bf16 v8bf;
typedef __attribute__((ext_vector_type(8)))  float  v8f;

// workspace layout (bytes), all 16B aligned
#define OFF_NBF  0u               // 2048*128 bf16  = 512 KB
#define OFF_SBF  524288u          // 2048*128 bf16  = 512 KB
#define OFF_W1T  1048576u         // 128*416 bf16   = 104 KB
#define OFF_W2T  1155072u         // 64*128  bf16   =  16 KB

// ---------------- prep: zero rho, bf16 atom features, transposed bf16 weights ----
__global__ __launch_bounds__(256) void prep_kernel(
    const float* __restrict__ S, const float* __restrict__ V,
    const float* __restrict__ W1, const float* __restrict__ W2,
    float* __restrict__ rho, char* __restrict__ ws)
{
  __bf16* nbf = (__bf16*)(ws + OFF_NBF);
  __bf16* sbf = (__bf16*)(ws + OFF_SBF);
  __bf16* w1t = (__bf16*)(ws + OFF_W1T);
  __bf16* w2t = (__bf16*)(ws + OFF_W2T);
  int tid = blockIdx.x * 256 + threadIdx.x;

  if (tid < B_ * P_) rho[tid] = 0.0f;

  if (tid < B_ * N_ * F_) {              // per atom-feature norms + bf16 S
    int a = tid / F_, f = tid % F_;
    float vx = V[a * 3 * F_ + 0 * F_ + f];
    float vy = V[a * 3 * F_ + 1 * F_ + f];
    float vz = V[a * 3 * F_ + 2 * F_ + f];
    nbf[tid] = (__bf16)sqrtf(vx * vx + vy * vy + vz * vz);
    sbf[tid] = (__bf16)S[tid];
  }
  if (tid < F_ * INDIM) {                // W1t[n][k] = W1[k][n]
    int n = tid / INDIM, k = tid % INDIM;
    w1t[n * INDIM + k] = (__bf16)W1[k * F_ + n];
  }
  if (tid < (F_ / 2) * F_) {             // W2t[n][k] = W2[k][n]
    int n = tid / F_, k = tid % F_;
    w2t[n * F_ + k] = (__bf16)W2[k * (F_ / 2) + n];
  }
}

// ---------------- main edge kernel: 1 wave/block, 32 edges (two 16-row A tiles) --
#define SH1 432   // h1 row stride (halfs): 416 + 16 pad, rows 16B aligned
#define SH2 136   // h2 row stride (halfs): 128 + 8 pad

__device__ __forceinline__ float silu_f(float x) {
  return x / (1.0f + __expf(-x));
}

__device__ __forceinline__ v16bf load_a_frag(const __bf16* row, int kb, int hs) {
  v8bf lo = *(const v8bf*)&row[kb + hs * 16];
  v8bf hi = *(const v8bf*)&row[kb + hs * 16 + 8];
  return __builtin_shufflevector(lo, hi, 0,1,2,3,4,5,6,7,8,9,10,11,12,13,14,15);
}

__device__ __forceinline__ v16bf load_b_frag(const __bf16* p) {
  v8bf lo = *(const v8bf*)p;
  v8bf hi = *(const v8bf*)(p + 8);
  return __builtin_shufflevector(lo, hi, 0,1,2,3,4,5,6,7,8,9,10,11,12,13,14,15);
}

__global__ __launch_bounds__(32) void edge_mlp_kernel(
    const float* __restrict__ V,    const float* __restrict__ axyz,
    const float* __restrict__ pxyz, const float* __restrict__ cell,
    const float* __restrict__ pdisp,
    const float* __restrict__ b1,   const float* __restrict__ b2,
    const float* __restrict__ W3,   const float* __restrict__ b3,
    const int*   __restrict__ pedge,
    const char*  __restrict__ ws,   float* __restrict__ rho)
{
  __shared__ __align__(16) __bf16 h1[32 * SH1];
  __shared__ __align__(16) __bf16 h2[32 * SH2];
  __shared__ float rhat [32][3];
  __shared__ float dists[32];
  __shared__ float cws  [32];
  __shared__ int   pidxs[32];
  __shared__ int   aidxs[32];

  const __bf16* nbf = (const __bf16*)(ws + OFF_NBF);
  const __bf16* sbf = (const __bf16*)(ws + OFF_SBF);
  const __bf16* w1t = (const __bf16*)(ws + OFF_W1T);
  const __bf16* w2t = (const __bf16*)(ws + OFF_W2T);

  const int l  = threadIdx.x & 31;
  const int m  = l & 15;        // column-lane within 16x16 tile
  const int hs = l >> 4;        // half-wave select
  const int e0 = blockIdx.x * 32;

  // ---- phase 1: per-edge geometry (all 32 lanes, one edge each) ----
  {
    int ge = e0 + l;
    int b  = ge / E_;
    int a  = b * N_ + pedge[ge * 2 + 0];
    int p  = b * P_ + pedge[ge * 2 + 1];
    float d0 = pdisp[ge * 3 + 0], d1 = pdisp[ge * 3 + 1], d2 = pdisp[ge * 3 + 2];
    const float* cb = cell + b * 9;
    float dpx = d0 * cb[0] + d1 * cb[3] + d2 * cb[6];
    float dpy = d0 * cb[1] + d1 * cb[4] + d2 * cb[7];
    float dpz = d0 * cb[2] + d1 * cb[5] + d2 * cb[8];
    float dx = pxyz[p * 3 + 0] - (axyz[a * 3 + 0] + dpx);
    float dy = pxyz[p * 3 + 1] - (axyz[a * 3 + 1] + dpy);
    float dz = pxyz[p * 3 + 2] - (axyz[a * 3 + 2] + dpz);
    float dist = sqrtf(dx * dx + dy * dy + dz * dz);
    float inv  = 1.0f / (dist + 1e-8f);
    rhat[l][0] = dx * inv; rhat[l][1] = dy * inv; rhat[l][2] = dz * inv;
    dists[l] = dist;
    cws[l]   = (dist < CUT) ? 0.5f * (__cosf(PI_ * dist / CUT) + 1.0f) : 0.0f;
    pidxs[l] = p;
    aidxs[l] = a;
  }
  __syncthreads();

  // ---- phase 2: build 32x416 bf16 feature tile in LDS ----
#pragma unroll
  for (int t = 0; t < 2; ++t) {
    int row = m + 16 * t;
    float dist = dists[row];
    float r0 = rhat[row][0], r1 = rhat[row][1], r2 = rhat[row][2];
    int   a  = aidxs[row];
    float invd = 1.0f / dist;
    // sinc RBF: this lane's columns hs*16 .. hs*16+15
#pragma unroll
    for (int j = 0; j < 16; ++j) {
      int c = hs * 16 + j;
      h1[row * SH1 + c] = (__bf16)(__sinf(dist * (float)(c + 1) * (PI_ / CUT)) * invd);
    }
    // q_pi (cols 32..159), n_e (160..287), S (288..415); this lane: f = hs*64..+63
    const float* Vb = V + a * 3 * F_;
    int f0 = hs * 64;
    for (int f = f0; f < f0 + 64; f += 8) {
      v8bf qv;
#pragma unroll
      for (int u = 0; u < 8; u += 4) {
        float4 vx = *(const float4*)(Vb + f + u);
        float4 vy = *(const float4*)(Vb + F_ + f + u);
        float4 vz = *(const float4*)(Vb + 2 * F_ + f + u);
        qv[u + 0] = (__bf16)(vx.x * r0 + vy.x * r1 + vz.x * r2);
        qv[u + 1] = (__bf16)(vx.y * r0 + vy.y * r1 + vz.y * r2);
        qv[u + 2] = (__bf16)(vx.z * r0 + vy.z * r1 + vz.z * r2);
        qv[u + 3] = (__bf16)(vx.w * r0 + vy.w * r1 + vz.w * r2);
      }
      *(v8bf*)&h1[row * SH1 + DRBF + f] = qv;
      *(v8bf*)&h1[row * SH1 + DRBF + F_ + f]     = *(const v8bf*)&nbf[a * F_ + f];
      *(v8bf*)&h1[row * SH1 + DRBF + 2 * F_ + f] = *(const v8bf*)&sbf[a * F_ + f];
    }
  }
  __syncthreads();

  // ---- phase 3: layer 1 GEMM 32x416 @ 416x128; B fragment shared by 2 A tiles ----
  v8f acc[2][8];
#pragma unroll
  for (int t = 0; t < 2; ++t)
#pragma unroll
    for (int j = 0; j < 8; ++j)
#pragma unroll
      for (int v = 0; v < 8; ++v) acc[t][j][v] = 0.0f;

  const __bf16* arow0 = &h1[m * SH1];
  const __bf16* arow1 = &h1[(m + 16) * SH1];
#pragma unroll 1
  for (int s = 0; s < 13; ++s) {
    int kb = s * 32;
    v16bf a0 = load_a_frag(arow0, kb, hs);
    v16bf a1 = load_a_frag(arow1, kb, hs);
#pragma unroll
    for (int j = 0; j < 8; ++j) {
      v16bf bv = load_b_frag(w1t + (j * 16 + m) * INDIM + kb + hs * 16);
      acc[0][j] = __builtin_amdgcn_wmma_f32_16x16x32_bf16(
          false, a0, false, bv, (short)0, acc[0][j], false, false);
      acc[1][j] = __builtin_amdgcn_wmma_f32_16x16x32_bf16(
          false, a1, false, bv, (short)0, acc[1][j], false, false);
    }
  }
  // bias + SiLU -> h2 (32x128 bf16)
#pragma unroll
  for (int j = 0; j < 8; ++j) {
    int col = j * 16 + m;
    float bb = b1[col];
#pragma unroll
    for (int t = 0; t < 2; ++t)
#pragma unroll
      for (int v = 0; v < 8; ++v) {
        float x = acc[t][j][v] + bb;
        h2[(v + 8 * hs + 16 * t) * SH2 + col] = (__bf16)silu_f(x);
      }
  }
  __syncthreads();

  // ---- phase 4: layer 2 GEMM 32x128 @ 128x64 ----
  v8f acc2[2][4];
#pragma unroll
  for (int t = 0; t < 2; ++t)
#pragma unroll
    for (int j = 0; j < 4; ++j)
#pragma unroll
      for (int v = 0; v < 8; ++v) acc2[t][j][v] = 0.0f;

  const __bf16* a2row0 = &h2[m * SH2];
  const __bf16* a2row1 = &h2[(m + 16) * SH2];
#pragma unroll
  for (int s = 0; s < 4; ++s) {
    int kb = s * 32;
    v16bf a0 = load_a_frag(a2row0, kb, hs);
    v16bf a1 = load_a_frag(a2row1, kb, hs);
#pragma unroll
    for (int j = 0; j < 4; ++j) {
      v16bf bv = load_b_frag(w2t + (j * 16 + m) * F_ + kb + hs * 16);
      acc2[0][j] = __builtin_amdgcn_wmma_f32_16x16x32_bf16(
          false, a0, false, bv, (short)0, acc2[0][j], false, false);
      acc2[1][j] = __builtin_amdgcn_wmma_f32_16x16x32_bf16(
          false, a1, false, bv, (short)0, acc2[1][j], false, false);
    }
  }

  // ---- phase 5: layer 3 dot (64 -> 1) + cutoff + scatter-add ----
  float part[2][8];
#pragma unroll
  for (int t = 0; t < 2; ++t)
#pragma unroll
    for (int v = 0; v < 8; ++v) part[t][v] = 0.0f;
#pragma unroll
  for (int j = 0; j < 4; ++j) {
    int col = j * 16 + m;
    float bb = b2[col];
    float wv = W3[col];
#pragma unroll
    for (int t = 0; t < 2; ++t)
#pragma unroll
      for (int v = 0; v < 8; ++v)
        part[t][v] += silu_f(acc2[t][j][v] + bb) * wv;
  }
  // reduce across the 16 lanes sharing each output row
#pragma unroll
  for (int off = 1; off < 16; off <<= 1)
#pragma unroll
    for (int t = 0; t < 2; ++t)
#pragma unroll
      for (int v = 0; v < 8; ++v)
        part[t][v] += __shfl_xor(part[t][v], off, 16);

  if (m == 0) {
    float b3v = b3[0];
#pragma unroll
    for (int t = 0; t < 2; ++t)
#pragma unroll
      for (int v = 0; v < 8; ++v) {
        int row = v + 8 * hs + 16 * t;
        float val = (part[t][v] + b3v) * cws[row];
        atomicAdd(&rho[pidxs[row]], val);   // global_atomic_add_f32 segment-sum
      }
  }
}

extern "C" void kernel_launch(void* const* d_in, const int* in_sizes, int n_in,
                              void* d_out, int out_size, void* d_ws, size_t ws_size,
                              hipStream_t stream) {
  const float* S     = (const float*)d_in[0];
  const float* V     = (const float*)d_in[1];
  const float* axyz  = (const float*)d_in[2];
  const float* pxyz  = (const float*)d_in[3];
  const float* cell  = (const float*)d_in[4];
  const float* pdisp = (const float*)d_in[5];
  const float* W1    = (const float*)d_in[6];
  const float* b1    = (const float*)d_in[7];
  const float* W2    = (const float*)d_in[8];
  const float* b2    = (const float*)d_in[9];
  const float* W3    = (const float*)d_in[10];
  const float* b3    = (const float*)d_in[11];
  const int*   pedge = (const int*)d_in[12];
  float* rho = (float*)d_out;
  char*  ws  = (char*)d_ws;

  int prep_threads = B_ * N_ * F_;      // 262144 covers all prep ranges
  prep_kernel<<<(prep_threads + 255) / 256, 256, 0, stream>>>(S, V, W1, W2, rho, ws);

  int blocks = (B_ * E_) / 32;          // 4096 blocks, 32 edges each, 1 wave/block
  edge_mlp_kernel<<<blocks, 32, 0, stream>>>(
      V, axyz, pxyz, cell, pdisp, b1, b2, W3, b3, pedge, ws, rho);
}
